// myModel_77034533421808
// MI455X (gfx1250) — matrix operand order repaired and checked by reading, not compiled, not run on hardware
//
#include <hip/hip_runtime.h>

// ---------------------------------------------------------------------------
// CDNA5 / gfx1250. wave32. All GEMMs via v_wmma_f32_16x16x32_f16 (f32 accum).
// Fragments are contiguous 16B chunks per the ISA VGPR layouts. Weights are
// pre-converted to f16, transposed [N][K], zero-padded. Bulk LDS<->HBM streams
// use the CDNA5 async-LDS path (ASYNCcnt + s_wait_asynccnt).
// ---------------------------------------------------------------------------

typedef __attribute__((ext_vector_type(16))) _Float16 v16h;
typedef __attribute__((ext_vector_type(8)))  _Float16 v8h;
typedef __attribute__((ext_vector_type(8)))  float    v8f;

#define H2 _Float16

// weight-pack offsets (halves)
#define OFF_W1  0        // [32 co][32 k]   (k>=27 zero)
#define OFF_W2  1024     // [64 co][288 k]
#define OFF_W3  19456    // [64 co][576 k]
#define OFF_SA1 56320    // [64 co][64 k]
#define OFF_SA2 60416
#define OFF_SA3 64512
#define OFF_SA4 68608    // [32 co][64 k]
#define OFF_CS1 70656    // [64 co][32 k]
#define OFF_CS2 72704
#define OFF_CS3 74752
#define OFF_CS4 76800    // [32 co][64 k]
#define OFF_CSS 78848    // [256 co][32 k]

__device__ __forceinline__ v8f wmma16x16x32(v16h a, v16h b, v8f c) {
  return __builtin_amdgcn_wmma_f32_16x16x32_f16(false, a, false, b, (short)0, c,
                                                false, false);
}

__device__ __forceinline__ v16h cat8(v8h lo, v8h hi) {
  union { v16h v; v8h h[2]; } u;
  u.h[0] = lo;
  u.h[1] = hi;
  return u.v;
}

__device__ __forceinline__ v16h load_a_rowmaj(const H2* base, int ld, int kk0,
                                              int lane) {
  const int m = lane & 15;
  const H2* p = base + (size_t)m * ld + kk0;
  const int kb = (lane >> 4) << 3;
  return cat8(*(const v8h*)(p + kb), *(const v8h*)(p + 16 + kb));
}

__device__ __forceinline__ v16h load_a_ptr(const H2* p, bool ok, int lane) {
  const int kb = (lane >> 4) << 3;
  v16h a = cat8(*(const v8h*)(p + kb), *(const v8h*)(p + 16 + kb));
  v16h z = {};
  return ok ? a : z;
}

__device__ __forceinline__ v16h load_b_colmaj(const H2* baseT, int ld, int n0,
                                              int kk0, int lane) {
  const int n = n0 + (lane & 15);
  const H2* p = baseT + (size_t)n * ld + kk0;
  const int kb = (lane >> 4) << 3;
  return cat8(*(const v8h*)(p + kb), *(const v8h*)(p + 16 + kb));
}

template <typename F>
__device__ __forceinline__ void store_d(int lane, const v8f& d, F out) {
  const int n  = lane & 15;
  const int mb = (lane >> 4) << 3;
#pragma unroll
  for (int r = 0; r < 8; ++r) out(mb + r, n, d[r]);
}

// ---- CDNA5 async LDS<->global helpers (ASYNCcnt) ---------------------------
// LDS operand: allocation-relative byte offset = low 32 bits of flat address.
__device__ __forceinline__ void async_g2l_b128(const H2* gaddr, const H2* lds) {
  const unsigned loff = (unsigned)(size_t)lds;
  asm volatile("global_load_async_to_lds_b128 %0, %1, off" ::"v"(loff),
               "v"(gaddr)
               : "memory");
}
__device__ __forceinline__ void async_l2g_b128(H2* gaddr, const H2* lds) {
  const unsigned loff = (unsigned)(size_t)lds;
  asm volatile("global_store_async_from_lds_b128 %0, %1, off" ::"v"(gaddr),
               "v"(loff)
               : "memory");
}
__device__ __forceinline__ void wait_async0() {
  asm volatile("s_wait_asynccnt 0" ::: "memory");
}

// ---------------------------------------------------------------------------
// Kernel 0a: x f32 -> f16 (so the backbone can async-DMA it straight to LDS)
// ---------------------------------------------------------------------------
__global__ __launch_bounds__(256) void k_prepx(const float* __restrict__ x,
                                               H2* __restrict__ xh) {
  const size_t i = (size_t)blockIdx.x * 256 + threadIdx.x;  // one v8h chunk
  const float4* x4 = (const float4*)x;
  const float4 a = x4[i * 2 + 0];
  const float4 b = x4[i * 2 + 1];
  v8h h = {(H2)a.x, (H2)a.y, (H2)a.z, (H2)a.w,
           (H2)b.x, (H2)b.y, (H2)b.z, (H2)b.w};
  *(v8h*)&xh[i * 8] = h;
}

// ---------------------------------------------------------------------------
// Kernel 0b: weight prep: f32 -> f16, transpose to [N][K], zero-pad conv1 K.
// ---------------------------------------------------------------------------
__global__ __launch_bounds__(256) void k_prep(
    const float* __restrict__ w1, const float* __restrict__ w2,
    const float* __restrict__ w3, const float* __restrict__ sa1,
    const float* __restrict__ sa2, const float* __restrict__ sa3,
    const float* __restrict__ sa4, const float* __restrict__ cs1,
    const float* __restrict__ cs2, const float* __restrict__ cs3,
    const float* __restrict__ cs4, const float* __restrict__ css,
    H2* __restrict__ wp) {
  const int t = blockIdx.x * 256 + threadIdx.x;
  const int stride = gridDim.x * 256;
  for (int i = t; i < 32 * 32; i += stride) {
    const int co = i >> 5, k = i & 31;
    wp[OFF_W1 + i] = (k < 27) ? (H2)w1[k * 32 + co] : (H2)0.f;
  }
  for (int i = t; i < 64 * 288; i += stride) {
    const int co = i / 288, k = i - co * 288;
    wp[OFF_W2 + i] = (H2)w2[k * 64 + co];
  }
  for (int i = t; i < 64 * 576; i += stride) {
    const int co = i / 576, k = i - co * 576;
    wp[OFF_W3 + i] = (H2)w3[k * 64 + co];
  }
  for (int i = t; i < 64 * 64; i += stride) {
    const int co = i >> 6, k = i & 63;
    wp[OFF_SA1 + i] = (H2)sa1[k * 64 + co];
    wp[OFF_SA2 + i] = (H2)sa2[k * 64 + co];
    wp[OFF_SA3 + i] = (H2)sa3[k * 64 + co];
  }
  for (int i = t; i < 32 * 64; i += stride) {
    const int co = i >> 6, k = i & 63;
    wp[OFF_SA4 + i] = (H2)sa4[k * 32 + co];
    wp[OFF_CS4 + i] = (H2)cs4[k * 32 + co];
  }
  for (int i = t; i < 64 * 32; i += stride) {
    const int co = i >> 5, k = i & 31;
    wp[OFF_CS1 + i] = (H2)cs1[k * 64 + co];
    wp[OFF_CS2 + i] = (H2)cs2[k * 64 + co];
    wp[OFF_CS3 + i] = (H2)cs3[k * 64 + co];
  }
  for (int i = t; i < 256 * 32; i += stride) {
    const int co = i >> 5, k = i & 31;
    wp[OFF_CSS + i] = (H2)css[k * 256 + co];
  }
}

// ---------------------------------------------------------------------------
// Kernel 1: fused CNN backbone, one image per workgroup, 156 KB LDS.
// Input tile arrives via async-DMA (no VGPR staging, no converts).
// ---------------------------------------------------------------------------
__global__ __launch_bounds__(256) void k_backbone(
    const H2* __restrict__ xh, const float* __restrict__ b1,
    const float* __restrict__ b2, const float* __restrict__ b3,
    const H2* __restrict__ wp, float* __restrict__ hg, H2* __restrict__ hh) {
  __shared__ H2    s_in[64 * 64 * 3];  // 24 KB
  __shared__ H2    s_c1[1024 * 40];    // 80 KB
  __shared__ H2    s_c2[256 * 72];     // 36 KB
  __shared__ float s_c3[64 * 64];      // 16 KB

  const int img  = blockIdx.x;
  const int tid  = threadIdx.x;
  const int wave = tid >> 5;
  const int lane = tid & 31;

  const H2* w1T = wp + OFF_W1;
  const H2* w2T = wp + OFF_W2;
  const H2* w3T = wp + OFF_W3;

  // async stage: 12288 halves = 768 x b128, 3 per thread
  {
    const H2* src = xh + (size_t)img * 12288;
#pragma unroll
    for (int c = 0; c < 3; ++c) {
      const int ch = tid + c * 256;
      async_g2l_b128(src + ch * 16, &s_in[ch * 16]);
    }
    wait_async0();
  }
  __syncthreads();

  // conv1: M=1024, N=32, K=27->32 (im2col gather, branchless)
  for (int job = wave; job < 128; job += 8) {
    const int mt = job >> 1, nt = job & 1;
    v16h bf = load_b_colmaj(w1T, 32, nt * 16, 0, lane);
    v16h af;
    {
      const int m  = lane & 15;
      const int kb = (lane >> 4) << 3;
      const int px = mt * 16 + m;
      const int oy = px >> 5, ox = px & 31;
#pragma unroll
      for (int i = 0; i < 16; ++i) {
        const int v = i >> 1, pp = i & 1;
        const int k = (v < 4) ? (kb + (v << 1) + pp)
                              : (16 + kb + ((v - 4) << 1) + pp);
        const int kc = (k < 27) ? k : 0;
        const int ky = kc / 9, r = kc - ky * 9;
        const int kx = r / 3, ci = r - kx * 3;
        const int iy = oy * 2 + ky, ix = ox * 2 + kx;
        const bool ok = (k < 27) && (iy < 64) && (ix < 64);
        const H2 ld = s_in[ok ? (iy * 64 + ix) * 3 + ci : 0];
        af[i] = ok ? ld : (H2)0.f;
      }
    }
    v8f acc = {};
    acc = wmma16x16x32(af, bf, acc);
    store_d(lane, acc, [&](int m, int n, float vv) {
      const int px = mt * 16 + m, co = nt * 16 + n;
      const float rr = vv + b1[co];
      s_c1[px * 40 + co] = (H2)(rr > 0.f ? rr : 0.f);
    });
  }
  __syncthreads();

  // conv2: M=256, N=64, K=288 = 9 taps x 32ch
  for (int job = wave; job < 64; job += 8) {
    const int mt = job >> 2, nt = job & 3;
    const int m  = lane & 15;
    const int px = mt * 16 + m;
    const int oy = px >> 4, ox = px & 15;
    v8f acc = {};
    for (int kp = 0; kp < 9; ++kp) {
      const int ky = kp / 3, kx = kp - ky * 3;
      const int iy = oy * 2 + ky, ix = ox * 2 + kx;
      const bool ok = (iy < 32) && (ix < 32);
      const H2* ap = s_c1 + (ok ? (iy * 32 + ix) : 0) * 40;
      v16h af = load_a_ptr(ap, ok, lane);
      v16h bf = load_b_colmaj(w2T, 288, nt * 16, kp * 32, lane);
      acc = wmma16x16x32(af, bf, acc);
    }
    store_d(lane, acc, [&](int mm, int n, float vv) {
      const int pxx = mt * 16 + mm, co = nt * 16 + n;
      const float rr = vv + b2[co];
      s_c2[pxx * 72 + co] = (H2)(rr > 0.f ? rr : 0.f);
    });
  }
  __syncthreads();

  // conv3: M=64, N=64, K=576 = 9 taps x 2 ch-blocks
  for (int job = wave; job < 16; job += 8) {
    const int mt = job >> 2, nt = job & 3;
    const int m  = lane & 15;
    const int px = mt * 16 + m;
    const int oy = px >> 3, ox = px & 7;
    v8f acc = {};
    for (int kp = 0; kp < 9; ++kp) {
      const int ky = kp / 3, kx = kp - ky * 3;
      const int iy = oy * 2 + ky, ix = ox * 2 + kx;
      const bool ok = (iy < 16) && (ix < 16);
      const H2* rowp = s_c2 + (ok ? (iy * 16 + ix) : 0) * 72;
      for (int cb = 0; cb < 2; ++cb) {
        v16h af = load_a_ptr(rowp + cb * 32, ok, lane);
        v16h bf = load_b_colmaj(w3T, 576, nt * 16, kp * 64 + cb * 32, lane);
        acc = wmma16x16x32(af, bf, acc);
      }
    }
    store_d(lane, acc, [&](int mm, int n, float vv) {
      const int pxx = mt * 16 + mm, co = nt * 16 + n;
      const float rr = vv + b3[co];
      s_c3[pxx * 64 + co] = rr > 0.f ? rr : 0.f;
    });
  }
  __syncthreads();

  // GAP 8x8
  if (tid < 64) {
    float s = 0.f;
    for (int p = 0; p < 64; ++p) s += s_c3[p * 64 + tid];
    s *= (1.f / 64.f);
    hg[(size_t)img * 64 + tid] = s;
    hh[(size_t)img * 64 + tid] = (H2)s;
  }
}

// ---------------------------------------------------------------------------
// Kernel 2: fc1 + softmax -> output1 [64,2]
// ---------------------------------------------------------------------------
__global__ __launch_bounds__(64) void k_fc1(const float* __restrict__ hg,
                                            const float* __restrict__ w,
                                            const float* __restrict__ b,
                                            float* __restrict__ out) {
  __shared__ float red[64][2];
  const int s = blockIdx.x, tid = threadIdx.x;
  float a0 = 0.f, a1 = 0.f;
  for (int i = tid; i < 1024; i += 64) {
    const float hv = hg[s * 1024 + i];
    a0 += hv * w[i * 2 + 0];
    a1 += hv * w[i * 2 + 1];
  }
  red[tid][0] = a0;
  red[tid][1] = a1;
  __syncthreads();
  for (int off = 32; off > 0; off >>= 1) {
    if (tid < off) {
      red[tid][0] += red[tid + off][0];
      red[tid][1] += red[tid + off][1];
    }
    __syncthreads();
  }
  if (tid == 0) {
    const float l0 = red[0][0] + b[0], l1 = red[0][1] + b[1];
    const float mx = fmaxf(l0, l1);
    const float e0 = __expf(l0 - mx), e1 = __expf(l1 - mx);
    const float inv = 1.f / (e0 + e1);
    out[s * 2 + 0] = e0 * inv;
    out[s * 2 + 1] = e1 * inv;
  }
}

// ---------------------------------------------------------------------------
// Kernel 3: self-attention set transform. One block per set (2 waves).
// ---------------------------------------------------------------------------
__global__ __launch_bounds__(64) void k_selfattn(const H2* __restrict__ hh,
                                                 const H2* __restrict__ wp,
                                                 H2* __restrict__ feat2) {
  __shared__ H2    s_f[16 * 72];
  __shared__ H2    s_q[16 * 72];
  __shared__ H2    s_k[16 * 72];
  __shared__ H2    s_vT[64 * 40];      // [channel][item], items 16..31 zero
  __shared__ H2    s_at[2 * 16 * 40];  // [h*16+n][m], m 16..31 zero
  __shared__ H2    s_o[16 * 72];
  __shared__ float s_sc[2 * 16 * 16];

  const int set = blockIdx.x, tid = threadIdx.x;
  const int wave = tid >> 5, lane = tid & 31;

  for (int i = tid; i < 128; i += 64) {
    const int m = i >> 3, c8 = (i & 7) * 8;
    *(v8h*)&s_f[m * 72 + c8] = *(const v8h*)&hh[set * 1024 + m * 64 + c8];
  }
  for (int i = tid; i < 64 * 40; i += 64) s_vT[i] = (H2)0.f;
  for (int i = tid; i < 2 * 16 * 40; i += 64) s_at[i] = (H2)0.f;
  __syncthreads();

  for (int job = wave; job < 12; job += 2) {
    const int proj = job >> 2, nt = job & 3;
    const H2* wT = wp + (proj == 0 ? OFF_SA1 : proj == 1 ? OFF_SA2 : OFF_SA3);
    H2* dst = (proj == 0) ? s_q : s_k;
    v8f acc = {};
    for (int ks = 0; ks < 2; ++ks) {
      v16h af = load_a_rowmaj(s_f, 72, ks * 32, lane);
      v16h bf = load_b_colmaj(wT, 64, nt * 16, ks * 32, lane);
      acc = wmma16x16x32(af, bf, acc);
    }
    store_d(lane, acc, [&](int m, int n, float vv) {
      const int col = nt * 16 + n;
      if (proj == 2) s_vT[col * 40 + m] = (H2)vv;
      else           dst[m * 72 + col] = (H2)vv;
    });
  }
  __syncthreads();

  {
    const int h = wave;
    v16h af = load_a_rowmaj(s_q, 72, h * 32, lane);
    v16h bf = load_b_colmaj(s_k, 72, 0, h * 32, lane);
    v8f acc = {};
    acc = wmma16x16x32(af, bf, acc);
    const float scale = 0.17677669529663688f;  // 1/sqrt(32)
    store_d(lane, acc, [&](int m, int n, float vv) {
      s_sc[(h * 16 + m) * 16 + n] = vv * scale;
    });
  }
  __syncthreads();

  if (tid < 32) {
    float* row = &s_sc[tid * 16];
    float mx = row[0];
    for (int i = 1; i < 16; ++i) mx = fmaxf(mx, row[i]);
    float e[16], sum = 0.f;
    for (int i = 0; i < 16; ++i) { e[i] = __expf(row[i] - mx); sum += e[i]; }
    const float inv = 1.f / sum;
    for (int i = 0; i < 16; ++i) s_at[tid * 40 + i] = (H2)(e[i] * inv);
  }
  __syncthreads();

  for (int job = wave; job < 4; job += 2) {
    const int h = job >> 1, nt = job & 1;
    v16h af = load_a_rowmaj(s_at + h * 16 * 40, 40, 0, lane);
    v16h bf = load_b_colmaj(s_vT, 40, h * 32 + nt * 16, 0, lane);
    v8f acc = {};
    acc = wmma16x16x32(af, bf, acc);
    store_d(lane, acc, [&](int m, int n, float vv) {
      s_o[m * 72 + h * 32 + nt * 16 + n] = (H2)vv;
    });
  }
  __syncthreads();

  for (int job = wave; job < 2; job += 2) {
    const int nt = job;
    v8f acc = {};
    for (int ks = 0; ks < 2; ++ks) {
      v16h af = load_a_rowmaj(s_o, 72, ks * 32, lane);
      v16h bf = load_b_colmaj(wp + OFF_SA4, 64, nt * 16, ks * 32, lane);
      acc = wmma16x16x32(af, bf, acc);
    }
    store_d(lane, acc, [&](int m, int n, float vv) {
      feat2[(set * 16 + m) * 32 + nt * 16 + n] = (H2)vv;
    });
  }
}

// ---------------------------------------------------------------------------
// Kernel 4: cseft q/k/v + softmax -> K-padded attn rows + transposed padded v
// ---------------------------------------------------------------------------
__global__ __launch_bounds__(64) void k_cseft(const H2* __restrict__ feat2,
                                              const H2* __restrict__ wp,
                                              H2* __restrict__ attnb,
                                              H2* __restrict__ vbufT) {
  __shared__ H2    s_f[16 * 40];
  __shared__ H2    s_q[16 * 72];
  __shared__ H2    s_k[16 * 72];
  __shared__ H2    s_v[16 * 72];
  __shared__ float s_sc[2 * 16 * 16];

  const int set = blockIdx.x, tid = threadIdx.x;
  const int wave = tid >> 5, lane = tid & 31;

  for (int i = tid; i < 64; i += 64) {
    const int m = i >> 2, c8 = (i & 3) * 8;
    *(v8h*)&s_f[m * 40 + c8] = *(const v8h*)&feat2[set * 512 + m * 32 + c8];
  }
  __syncthreads();

  for (int job = wave; job < 12; job += 2) {
    const int proj = job >> 2, nt = job & 3;
    const H2* wT = wp + (proj == 0 ? OFF_CS1 : proj == 1 ? OFF_CS2 : OFF_CS3);
    H2* dst = (proj == 0) ? s_q : (proj == 1) ? s_k : s_v;
    v16h af = load_a_rowmaj(s_f, 40, 0, lane);
    v16h bf = load_b_colmaj(wT, 32, nt * 16, 0, lane);
    v8f acc = {};
    acc = wmma16x16x32(af, bf, acc);
    store_d(lane, acc, [&](int m, int n, float vv) {
      dst[m * 72 + nt * 16 + n] = (H2)vv;
    });
  }
  __syncthreads();

  {
    const int h = wave;
    v16h af = load_a_rowmaj(s_q, 72, h * 32, lane);
    v16h bf = load_b_colmaj(s_k, 72, 0, h * 32, lane);
    v8f acc = {};
    acc = wmma16x16x32(af, bf, acc);
    const float scale = 0.17677669529663688f;
    store_d(lane, acc, [&](int m, int n, float vv) {
      s_sc[(h * 16 + m) * 16 + n] = vv * scale;
    });
  }
  __syncthreads();

  if (tid < 32) {
    float* row = &s_sc[tid * 16];
    float mx = row[0];
    for (int i = 1; i < 16; ++i) mx = fmaxf(mx, row[i]);
    float e[16], sum = 0.f;
    for (int i = 0; i < 16; ++i) { e[i] = __expf(row[i] - mx); sum += e[i]; }
    const float inv = 1.f / sum;
    const int h = tid >> 4, n = tid & 15;
    H2* arow = attnb + ((size_t)(set * 2 + h) * 16 + n) * 32;
    for (int i = 0; i < 32; ++i)
      arow[i] = (i < 16) ? (H2)(e[i] * inv) : (H2)0.f;
  }
  for (int i = tid; i < 2048; i += 64) {
    const int h = i >> 10, rem = i & 1023;
    const int d = rem >> 5, m = rem & 31;
    const H2 val = (m < 16) ? s_v[m * 72 + h * 32 + d] : (H2)0.f;
    vbufT[((size_t)(set * 2 + h) * 32 + d) * 32 + m] = val;
  }
}

// ---------------------------------------------------------------------------
// Kernel 5: per pair (j,i): weighted -> pair -> z; z streamed out of LDS with
// async stores (ASYNCcnt).
// ---------------------------------------------------------------------------
__global__ __launch_bounds__(64) void k_pairz(const H2* __restrict__ attnb,
                                              const H2* __restrict__ vbufT,
                                              const H2* __restrict__ wp,
                                              H2* __restrict__ zbuf) {
  __shared__ H2 s_w[16 * 72];
  __shared__ H2 s_p[16 * 40];
  __shared__ H2 s_z[16 * 264];
  const int j = blockIdx.x >> 6, i = blockIdx.x & 63;
  const int tid = threadIdx.x, wave = tid >> 5, lane = tid & 31;

  for (int job = wave; job < 4; job += 2) {
    const int h = job >> 1, nt = job & 1;
    v16h af = load_a_rowmaj(attnb + (size_t)(i * 2 + h) * 16 * 32, 32, 0, lane);
    v16h bf = load_b_colmaj(vbufT + (size_t)(j * 2 + h) * 32 * 32, 32, nt * 16,
                            0, lane);
    v8f acc = {};
    acc = wmma16x16x32(af, bf, acc);
    store_d(lane, acc, [&](int m, int n, float vv) {
      s_w[m * 72 + h * 32 + nt * 16 + n] = (H2)vv;
    });
  }
  __syncthreads();

  for (int job = wave; job < 2; job += 2) {
    const int nt = job;
    v8f acc = {};
    for (int ks = 0; ks < 2; ++ks) {
      v16h af = load_a_rowmaj(s_w, 72, ks * 32, lane);
      v16h bf = load_b_colmaj(wp + OFF_CS4, 64, nt * 16, ks * 32, lane);
      acc = wmma16x16x32(af, bf, acc);
    }
    store_d(lane, acc,
            [&](int m, int n, float vv) { s_p[m * 40 + nt * 16 + n] = (H2)vv; });
  }
  __syncthreads();

  for (int job = wave; job < 16; job += 2) {
    const int nt = job;
    v16h af = load_a_rowmaj(s_p, 40, 0, lane);
    v16h bf = load_b_colmaj(wp + OFF_CSS, 32, nt * 16, 0, lane);
    v8f acc = {};
    acc = wmma16x16x32(af, bf, acc);
    store_d(lane, acc, [&](int m, int n, float vv) {
      s_z[m * 264 + nt * 16 + n] = (H2)vv;
    });
  }
  __syncthreads();

  // z tile out: 4096 halves = 256 x b128 async stores (4 per thread)
  {
    H2* dst = zbuf + (size_t)blockIdx.x * 16 * 256;
#pragma unroll
    for (int c = 0; c < 4; ++c) {
      const int ch = tid + c * 64;          // 0..255
      const int m = ch >> 4, c16 = (ch & 15) * 16;
      async_l2g_b128(dst + (size_t)m * 256 + c16, &s_z[m * 264 + c16]);
    }
    wait_async0();
  }
}

// ---------------------------------------------------------------------------
// Kernel 6: score[j,i] via z(j,i,h) z(i,j,h)^T, relu-mean, head mix.
// ---------------------------------------------------------------------------
__global__ __launch_bounds__(64) void k_score(const H2* __restrict__ zbuf,
                                              const float* __restrict__ w2,
                                              float* __restrict__ score) {
  __shared__ float red[64];
  const int j = blockIdx.x >> 6, i = blockIdx.x & 63;
  const int tid = threadIdx.x, wave = tid >> 5, lane = tid & 31;
  const int h = wave;
  const H2* zji = zbuf + (size_t)(j * 64 + i) * 16 * 256;
  const H2* zij = zbuf + (size_t)(i * 64 + j) * 16 * 256;

  v8f acc = {};
  for (int ks = 0; ks < 4; ++ks) {
    v16h af = load_a_rowmaj(zji, 256, h * 128 + ks * 32, lane);
    v16h bf = load_b_colmaj(zij, 256, 0, h * 128 + ks * 32, lane);
    acc = wmma16x16x32(af, bf, acc);
  }
  const float scale = 0.08838834764831845f;  // 1/sqrt(128)
  float part = 0.f;
#pragma unroll
  for (int r = 0; r < 8; ++r) {
    const float vv = acc[r] * scale;
    part += vv > 0.f ? vv : 0.f;
  }
  red[tid] = part;
  __syncthreads();
  if (tid == 0) {
    float s0 = 0.f, s1 = 0.f;
    for (int t = 0; t < 32; ++t) { s0 += red[t]; s1 += red[t + 32]; }
    score[j * 64 + i] =
        (s0 * (1.f / 256.f)) * w2[0] + (s1 * (1.f / 256.f)) * w2[1];
  }
}

// ---------------------------------------------------------------------------
// Kernel 7: matching head
// ---------------------------------------------------------------------------
__global__ __launch_bounds__(256) void k_head(
    const float* __restrict__ score, const float* __restrict__ fc2w,
    const float* __restrict__ fc2b, const float* __restrict__ fc3w,
    const float* __restrict__ fc3b, float* __restrict__ out2) {
  const int p = blockIdx.x * 256 + threadIdx.x;
  if (p >= 4096) return;
  const float sc = score[p];
  float l0 = fc3b[0], l1 = fc3b[1];
  for (int c = 0; c < 32; ++c) {
    float f = sc * fc2w[c] + fc2b[c];
    f = f > 0.f ? f : 0.f;
    l0 += f * fc3w[c * 2 + 0];
    l1 += f * fc3w[c * 2 + 1];
  }
  const float mx = fmaxf(l0, l1);
  const float e0 = __expf(l0 - mx), e1 = __expf(l1 - mx);
  const float inv = 1.f / (e0 + e1);
  out2[p * 2 + 0] = e0 * inv;
  out2[p * 2 + 1] = e1 * inv;
}

// ---------------------------------------------------------------------------
extern "C" void kernel_launch(void* const* d_in, const int* in_sizes, int n_in,
                              void* d_out, int out_size, void* d_ws,
                              size_t ws_size, hipStream_t stream) {
  (void)in_sizes; (void)n_in; (void)out_size; (void)ws_size;
  const float* x     = (const float*)d_in[0];
  const float* c1w   = (const float*)d_in[1];
  const float* c1b   = (const float*)d_in[2];
  const float* c2w   = (const float*)d_in[3];
  const float* c2b   = (const float*)d_in[4];
  const float* c3w   = (const float*)d_in[5];
  const float* c3b   = (const float*)d_in[6];
  const float* fc1w  = (const float*)d_in[7];
  const float* fc1b  = (const float*)d_in[8];
  const float* saw1  = (const float*)d_in[9];
  const float* saw2  = (const float*)d_in[10];
  const float* saw3  = (const float*)d_in[11];
  const float* saw4  = (const float*)d_in[12];
  const float* csw1  = (const float*)d_in[13];
  const float* csw2  = (const float*)d_in[14];
  const float* csw3  = (const float*)d_in[15];
  const float* csw4  = (const float*)d_in[16];
  const float* cssw  = (const float*)d_in[17];
  const float* cssw2 = (const float*)d_in[18];
  const float* fc2w  = (const float*)d_in[19];
  const float* fc2b  = (const float*)d_in[20];
  const float* fc3w  = (const float*)d_in[21];
  const float* fc3b  = (const float*)d_in[22];

  char* ws = (char*)d_ws;
  H2*    wp     = (H2*)(ws + 0);            // 174080 B f16 weight pack
  float* hg     = (float*)(ws + 174080);    // 1024*64 f32
  H2*    hh     = (H2*)(ws + 436224);       // 1024*64 f16
  H2*    feat2  = (H2*)(ws + 567296);       // 1024*32 f16
  H2*    attnb  = (H2*)(ws + 632832);       // 64*2*16*32 f16 (K-padded)
  H2*    vbufT  = (H2*)(ws + 763904);       // 64*2*32*32 f16
  float* score  = (float*)(ws + 1026048);   // 4096 f32
  H2*    zbuf   = (H2*)(ws + 1042432);      // 4096*16*256 f16 = 32 MB
  H2*    xh     = (H2*)(ws + 34596864);     // 1024*12288 f16 = 24 MB

  float* out = (float*)d_out;  // [4160,2]

  k_prepx<<<6144, 256, 0, stream>>>(x, xh);
  k_prep<<<64, 256, 0, stream>>>(c1w, c2w, c3w, saw1, saw2, saw3, saw4, csw1,
                                 csw2, csw3, csw4, cssw, wp);
  k_backbone<<<1024, 256, 0, stream>>>(xh, c1b, c2b, c3b, wp, hg, hh);
  k_fc1<<<64, 64, 0, stream>>>(hg, fc1w, fc1b, out);
  k_selfattn<<<64, 64, 0, stream>>>(hh, wp, feat2);
  k_cseft<<<64, 64, 0, stream>>>(feat2, wp, attnb, vbufT);
  k_pairz<<<4096, 64, 0, stream>>>(attnb, vbufT, wp, zbuf);
  k_score<<<4096, 64, 0, stream>>>(zbuf, cssw2, score);
  k_head<<<16, 256, 0, stream>>>(score, fc2w, fc2b, fc3w, fc3b, out + 128);
}
